// WaterGNN_7361573945453
// MI455X (gfx1250) — matrix operand order repaired and checked by reading, not compiled, not run on hardware
//
#include <hip/hip_runtime.h>
#include <hip/hip_bf16.h>
#include <stdint.h>

typedef __attribute__((ext_vector_type(16))) _Float16 v16h;
typedef __attribute__((ext_vector_type(8)))  float    v8f;

#define NODE_C_IN 128   // K for both layers

#if defined(__HIP_DEVICE_COMPILE__) && defined(__gfx1250__)
#define HAS_ASYNC_LDS 1
#else
#define HAS_ASYNC_LDS 0
#endif

// ---------------------------------------------------------------------------
// Degree / normalization
// ---------------------------------------------------------------------------
__global__ void fill_deg_kernel(float* __restrict__ deg, int n) {
    int i = blockIdx.x * blockDim.x + threadIdx.x;
    if (i < n) deg[i] = 1.0f;  // self-loop contributes 1 to in-degree
}

__global__ void degree_kernel(const long long* __restrict__ dst,
                              float* __restrict__ deg, int nE) {
    int e = blockIdx.x * blockDim.x + threadIdx.x;
    if (e < nE) {
        long long d = dst[e];
        atomicAdd(&deg[d], 1.0f);
    }
}

__global__ void dinv_kernel(const float* __restrict__ deg,
                            float* __restrict__ dinv, int n) {
    int i = blockIdx.x * blockDim.x + threadIdx.x;
    if (i < n) {
        float dg = deg[i];
        dinv[i] = (dg > 0.0f) ? rsqrtf(dg) : 0.0f;
    }
}

// ---------------------------------------------------------------------------
// Pack W (f32, [128 x OUTC] row-major) into f16 WMMA B-fragment order:
//   Wpk[step*512 + lane*16 + i],  step = (k0/32)*nColTiles + colTile
//   lane half 0: K = k0 + i,  half 1: K = k0 + 16 + i;  N = colTile*16 + lane%16
// Each lane's 16-half fragment is then a contiguous 32-byte chunk.
// ---------------------------------------------------------------------------
template <int OUTC>
__global__ void pack_w_kernel(const float* __restrict__ W,
                              _Float16* __restrict__ Wpk) {
    constexpr int nColTiles = OUTC / 16;
    const int idx = blockIdx.x * blockDim.x + threadIdx.x;   // over 128*OUTC
    if (idx >= NODE_C_IN * OUTC) return;
    const int step    = idx >> 9;       // /512
    const int within  = idx & 511;
    const int lane    = within >> 4;
    const int i       = within & 15;
    const int colTile = step % nColTiles;
    const int kt      = step / nColTiles;
    const int half    = lane >> 4;
    const int mLane   = lane & 15;
    const int k = kt * 32 + half * 16 + i;
    const int n = colTile * 16 + mLane;
    Wpk[idx] = (_Float16)W[(size_t)k * OUTC + n];
}

// ---------------------------------------------------------------------------
// WMMA GEMM:  H[n x OUTC] = (RELU_IN ? relu(X) : X)[n x 128] @ W[128 x OUTC]
// Packed-f16 W is staged once per block into LDS via gfx1250 async-to-LDS
// (ASYNCcnt-tracked), then each wave computes one 16x16 tile, K-loop 4 x 32.
// A and B fragments both use contiguous K order, K = 16*half + i (wave32).
// ---------------------------------------------------------------------------
template <int OUTC, bool RELU_IN>
__global__ void gemm_wmma_kernel(const float* __restrict__ X,
                                 const _Float16* __restrict__ Wpk,
                                 float* __restrict__ H, int nRows) {
    constexpr int KDIM       = NODE_C_IN;
    constexpr int nColTiles  = OUTC / 16;       // 8 (layer1) or 4 (layer2)
    constexpr int rowsPerBlk = 8 / nColTiles;   // 8 waves per 256-thread block

    __shared__ __align__(32) _Float16 wlds[KDIM * OUTC];

    // ---- stage packed W into LDS (32KB / 16KB), 16B per thread per step ----
    {
        constexpr int totalBytes = KDIM * OUTC * 2;
        const char* g = (const char*)Wpk;
        char* l = (char*)wlds;
        for (int off = threadIdx.x * 16; off < totalBytes; off += 256 * 16) {
#if HAS_ASYNC_LDS
            // LDS byte address = low 32 bits of generic LDS pointer
            // (generic-LDS aperture: addr[31:0] is the LDS offset).
            const uint32_t ldsAddr = (uint32_t)(uintptr_t)(l + off);
            const uint64_t gAddr   = (uint64_t)(uintptr_t)(g + off);
            asm volatile("global_load_async_to_lds_b128 %0, %1, off"
                         :: "v"(ldsAddr), "v"(gAddr)
                         : "memory");
#else
            *(float4*)(l + off) = *(const float4*)(g + off);
#endif
        }
#if HAS_ASYNC_LDS
        asm volatile("s_wait_asynccnt 0" ::: "memory");
#endif
        __syncthreads();   // every thread must reach this: no early returns above
    }

    const int lane = threadIdx.x & 31;
    const int wave = threadIdx.x >> 5;

    const int rowTile = blockIdx.x * rowsPerBlk + wave / nColTiles;
    const int colTile = wave % nColTiles;
    const int m0 = rowTile * 16;
    if (m0 < nRows) {
        const int mLane = lane & 15;
        const int half  = lane >> 4;            // 0: lanes 0-15, 1: lanes 16-31

        int aRow = m0 + mLane;
        if (aRow >= nRows) aRow = nRows - 1;    // clamp for ragged tail
        const float* __restrict__ xrow = X + (size_t)aRow * KDIM;
        const int kbA = half * 16;              // A: lanes 16-31 start at K=16

        v8f c = {};
        #pragma unroll
        for (int kt = 0; kt < KDIM / 32; ++kt) {
            // ---- A fragment: 16x32 f16, one M-row per lane ----
            v16h a;
            #pragma unroll
            for (int i = 0; i < 16; i += 2) {
                const int k = kbA + i;  // even -> float2 aligned
                float2 f = *(const float2*)(xrow + kt * 32 + k);
                if (RELU_IN) { f.x = fmaxf(f.x, 0.0f); f.y = fmaxf(f.y, 0.0f); }
                a[i]     = (_Float16)f.x;
                a[i + 1] = (_Float16)f.y;
            }
            // ---- B fragment: contiguous 32B per lane from LDS ----
            const int step = kt * nColTiles + colTile;
            v16h b = *(const v16h*)&wlds[(size_t)step * 512 + lane * 16];

            c = __builtin_amdgcn_wmma_f32_16x16x32_f16(
                    /*neg_a=*/false, a, /*neg_b=*/false, b,
                    /*c_mod=*/(short)0, c, /*reuse_a=*/false, /*reuse_b=*/false);
        }

        // ---- C/D: VGPR r holds M = m0 + 8*half + r, N = n0 + (lane&15) ----
        const int mBase = m0 + half * 8;
        const int n0 = colTile * 16;
        #pragma unroll
        for (int r = 0; r < 8; ++r) {
            const int m = mBase + r;
            if (m < nRows) H[(size_t)m * OUTC + (n0 + mLane)] = c[r];
        }
    }
}

// ---------------------------------------------------------------------------
// OUT[i][c] = bias[c] + dinv[i]^2 * H[i][c]   (bias + self-loop term fused)
// ---------------------------------------------------------------------------
template <int C>
__global__ void init_out_kernel(const float* __restrict__ H,
                                const float* __restrict__ bias,
                                const float* __restrict__ dinv,
                                float* __restrict__ OUT, int n) {
    size_t idx = (size_t)blockIdx.x * blockDim.x + threadIdx.x;
    if (idx >= (size_t)n * C) return;
    const int i = (int)(idx / C);
    const int ch = (int)(idx % C);
    const float di = dinv[i];
    OUT[idx] = bias[ch] + di * di * H[idx];
}

// ---------------------------------------------------------------------------
// Edge aggregation: OUT[dst] += dinv[src]*dinv[dst] * H[src]
// One wave per edge; channels striped lane + 32*j -> fully coalesced
// 128B load + 128B atomic transactions. Pure HBM-streaming kernel.
// ---------------------------------------------------------------------------
template <int C>
__global__ void edge_agg_kernel(const long long* __restrict__ src,
                                const long long* __restrict__ dst,
                                const float* __restrict__ H,
                                const float* __restrict__ dinv,
                                float* __restrict__ OUT, int nE) {
    const int wavesPerBlk = blockDim.x >> 5;
    const int e = blockIdx.x * wavesPerBlk + (threadIdx.x >> 5);
    if (e >= nE) return;
    const int lane = threadIdx.x & 31;

    if (lane == 0) {  // stream the index arrays ahead (global_prefetch_b8)
        __builtin_prefetch(src + e + 256, 0, 0);
        __builtin_prefetch(dst + e + 256, 0, 0);
    }

    const long long s = src[e];
    const long long d = dst[e];
    const float norm = dinv[s] * dinv[d];
    const float* __restrict__ hs = H + (size_t)s * C + lane;
    float* __restrict__ od = OUT + (size_t)d * C + lane;
    #pragma unroll
    for (int j = 0; j < C / 32; ++j)
        atomicAdd(od + (size_t)j * 32, norm * hs[(size_t)j * 32]);
}

// ---------------------------------------------------------------------------
// Host-side launch
// ---------------------------------------------------------------------------
static inline size_t alignUp256(size_t v) { return (v + 255) & ~(size_t)255; }

extern "C" void kernel_launch(void* const* d_in, const int* in_sizes, int n_in,
                              void* d_out, int out_size, void* d_ws, size_t ws_size,
                              hipStream_t stream) {
    const float*     x    = (const float*)d_in[0];
    const long long* eidx = (const long long*)d_in[1];   // (2, E) int64
    const float*     W1   = (const float*)d_in[2];
    const float*     b1   = (const float*)d_in[3];
    const float*     W2   = (const float*)d_in[4];
    const float*     b2   = (const float*)d_in[5];
    float* out = (float*)d_out;

    const int N  = in_sizes[0] / NODE_C_IN;   // 100000
    const int nE = in_sizes[1] / 2;           // 3200000
    const long long* src = eidx;              // edge_index[0]
    const long long* dst = eidx + nE;         // edge_index[1]

    // Workspace carve-out
    char* ws = (char*)d_ws;
    float*     deg  = (float*)ws;     ws += alignUp256((size_t)N * 4);
    float*     dinv = (float*)ws;     ws += alignUp256((size_t)N * 4);
    _Float16*  Wpk1 = (_Float16*)ws;  ws += alignUp256((size_t)NODE_C_IN * 128 * 2);
    _Float16*  Wpk2 = (_Float16*)ws;  ws += alignUp256((size_t)NODE_C_IN * 64 * 2);
    float*     h    = (float*)ws;     ws += alignUp256((size_t)N * 128 * 4);
    float*     out1 = (float*)ws;     ws += alignUp256((size_t)N * 128 * 4);
    float*     g    = h;  // layer-2 GEMM output reuses h's buffer (h dead by then)

    const int TPB = 256;

    // 0) pack weights to f16 fragment order (tiny)
    pack_w_kernel<128><<<(NODE_C_IN * 128 + TPB - 1) / TPB, TPB, 0, stream>>>(W1, Wpk1);
    pack_w_kernel<64><<<(NODE_C_IN * 64 + TPB - 1) / TPB, TPB, 0, stream>>>(W2, Wpk2);

    // 1) degrees (with self-loops) and dinv
    fill_deg_kernel<<<(N + TPB - 1) / TPB, TPB, 0, stream>>>(deg, N);
    degree_kernel<<<(nE + TPB - 1) / TPB, TPB, 0, stream>>>(dst, deg, nE);
    dinv_kernel<<<(N + TPB - 1) / TPB, TPB, 0, stream>>>(deg, dinv, N);

    // 2) layer 1: h = x @ W1  (WMMA), out1 = b1 + selfloop + edge scatter
    const int rowTiles = (N + 15) / 16;
    gemm_wmma_kernel<128, false><<<rowTiles, TPB, 0, stream>>>(x, Wpk1, h, N);
    {
        size_t total = (size_t)N * 128;
        init_out_kernel<128><<<(unsigned)((total + TPB - 1) / TPB), TPB, 0, stream>>>(
            h, b1, dinv, out1, N);
    }
    edge_agg_kernel<128><<<(nE + 7) / 8, TPB, 0, stream>>>(src, dst, h, dinv, out1, nE);

    // 3) layer 2: g = relu(out1) @ W2 (ReLU fused into A-fragment load)
    gemm_wmma_kernel<64, true><<<(rowTiles + 1) / 2, TPB, 0, stream>>>(out1, Wpk2, g, N);
    {
        size_t total = (size_t)N * 64;
        init_out_kernel<64><<<(unsigned)((total + TPB - 1) / TPB), TPB, 0, stream>>>(
            g, b2, dinv, out, N);
    }
    edge_agg_kernel<64><<<(nE + 7) / 8, TPB, 0, stream>>>(src, dst, g, dinv, out, nE);
}
